// GNN_SAGE_11665131176188
// MI455X (gfx1250) — matrix operand order repaired and checked by reading, not compiled, not run on hardware
//
#include <hip/hip_runtime.h>
#include <hip/hip_bf16.h>

#define NN 50000
#define NE 600000
#define DIM 128
#define NG 500

typedef __attribute__((ext_vector_type(16))) __bf16 v16bf;
typedef __attribute__((ext_vector_type(8)))  __bf16 v8bf;
typedef __attribute__((ext_vector_type(8)))  float  v8f;

// ---------------- utility: zero a float buffer ----------------
__global__ void k_zero(float* __restrict__ p, int n) {
    int i = blockIdx.x * blockDim.x + threadIdx.x;
    if (i < n) p[i] = 0.0f;
}

// ---------------- in-degree (float) ----------------
__global__ void k_degree(const int* __restrict__ dst, float* __restrict__ cnt) {
    int e = blockIdx.x * blockDim.x + threadIdx.x;
    if (e < NE) atomicAdd(&cnt[dst[e]], 1.0f);
}

// ---------------- edge gather + scatter-add --------------------
// one thread per (edge, 4-feature quad): float4 load + 4 f32 atomics (L2-resident agg)
__global__ void k_scatter(const float* __restrict__ x, const int* __restrict__ src,
                          const int* __restrict__ dst, float* __restrict__ agg) {
    long long t = (long long)blockIdx.x * blockDim.x + threadIdx.x;
    int e = (int)(t >> 5);
    if (e >= NE) return;
    int q = ((int)t & 31) << 2;
    int s = src[e], d = dst[e];
    const float4 v = *(const float4*)(x + (size_t)s * DIM + q);
    float* o = agg + (size_t)d * DIM + q;
    atomicAdd(o + 0, v.x); atomicAdd(o + 1, v.y);
    atomicAdd(o + 2, v.z); atomicAdd(o + 3, v.w);
}

// ---------------- weight pre-pack into WMMA B-fragment order ----------------
// Wcat = [Wl ; Wr] (256x128 f32) -> bf16 fragments [kt(8)][nt(8)][lane(32)][e(16)]
// B pack (32x16 bf16, this code's contiguous k order, matched by A): lanes 0-15: N=lane, K=k0+e ; lanes 16-31: N=lane-16, K=k0+16+e
__global__ void k_pack(const float* __restrict__ Wl, const float* __restrict__ Wr,
                       __bf16* __restrict__ out) {
    int i = blockIdx.x * blockDim.x + threadIdx.x;   // 0 .. 32767
    if (i >= 8 * 8 * 32 * 16) return;
    int e    =  i        & 15;
    int lane = (i >> 4)  & 31;
    int nt   = (i >> 9)  & 7;
    int kt   = (i >> 12) & 7;
    int k = kt * 32 + ((lane >> 4) << 4) + e;
    int n = nt * 16 + (lane & 15);
    float v = (k < 128) ? Wl[k * 128 + n] : Wr[(k - 128) * 128 + n];
    out[i] = (__bf16)v;
}

// ---------------- fused SAGE layer GEMM ----------------
// out = relu( [agg/cnt | x] @ [Wl;Wr] + bias ), one 16-row M-tile per block,
// 4 waves x 2 N-tiles each, K=256 in 8 steps of v_wmma_f32_16x16x32_bf16.
__global__ __launch_bounds__(128) void k_sage_gemm(
    const float* __restrict__ x, const float* __restrict__ agg,
    const float* __restrict__ cnt, const __bf16* __restrict__ wpack,
    const float* __restrict__ bias, float* __restrict__ out) {

    // padded LDS A-tile: 16 rows x 256 bf16, row stride 264 (528B -> bank spread, 16B aligned)
    __shared__ __attribute__((aligned(16))) __bf16 As[16 * 264];

    const int m0  = blockIdx.x * 16;
    const int tid = threadIdx.x;

    // stage: cols 0..127 = normalized aggregate, 128..255 = root features
    for (int i = tid; i < 16 * 128; i += 128) {
        int r = i >> 7, c = i & 127;
        size_t g = (size_t)(m0 + r) * DIM + c;
        float inv = 1.0f / fmaxf(cnt[m0 + r], 1.0f);
        As[r * 264 + c]       = (__bf16)(agg[g] * inv);
        As[r * 264 + 128 + c] = (__bf16)x[g];
    }
    __syncthreads();

    const int wv   = tid >> 5;
    const int lane = tid & 31;
    const int m    = lane & 15;    // A row / C column-in-tile
    const int h    = lane >> 4;    // lane half

    v8f acc0 = {}, acc1 = {};
    const v16bf* bfrags = (const v16bf*)wpack;

    for (int kt = 0; kt < 8; ++kt) {
        // A fragment (contiguous, matches B pack): K = 16h+e
        const __bf16* arow = &As[m * 264 + kt * 32 + 16 * h];
        v8bf a_lo = *(const v8bf*)(arow);
        v8bf a_hi = *(const v8bf*)(arow + 8);
        v16bf afrag;
#pragma unroll
        for (int e = 0; e < 8; ++e) { afrag[e] = a_lo[e]; afrag[e + 8] = a_hi[e]; }

        v16bf b0 = bfrags[(kt * 8 + (2 * wv + 0)) * 32 + lane];
        v16bf b1 = bfrags[(kt * 8 + (2 * wv + 1)) * 32 + lane];

        acc0 = __builtin_amdgcn_wmma_f32_16x16x32_bf16(false, afrag, false, b0,
                                                       (short)0, acc0, false, false);
        acc1 = __builtin_amdgcn_wmma_f32_16x16x32_bf16(false, afrag, false, b1,
                                                       (short)0, acc1, false, false);
    }

    // epilogue: C/D layout -> lane half h gives rows r+8h, column = nt*16 + m
#pragma unroll
    for (int t = 0; t < 2; ++t) {
        v8f acc = t ? acc1 : acc0;
        int n = (2 * wv + t) * 16 + m;
        float b = bias[n];
#pragma unroll
        for (int r = 0; r < 8; ++r) {
            float v = acc[r] + b;
            out[(size_t)(m0 + r + 8 * h) * DIM + n] = fmaxf(v, 0.0f);
        }
    }
}

// ---------------- graph mean-pool scatter ----------------
__global__ void k_pool(const float* __restrict__ x, const int* __restrict__ batch,
                       float* __restrict__ gsum, float* __restrict__ gcnt) {
    long long t = (long long)blockIdx.x * blockDim.x + threadIdx.x;
    int n = (int)(t >> 5);
    if (n >= NN) return;
    int q = ((int)t & 31) << 2;
    int b = batch[n];
    const float4 v = *(const float4*)(x + (size_t)n * DIM + q);
    float* o = gsum + (size_t)b * DIM + q;
    atomicAdd(o + 0, v.x); atomicAdd(o + 1, v.y);
    atomicAdd(o + 2, v.z); atomicAdd(o + 3, v.w);
    if (q == 0) atomicAdd(&gcnt[b], 1.0f);
}

// ---------------- final projection ----------------
__global__ void k_out(const float* __restrict__ gsum, const float* __restrict__ gcnt,
                      const float* __restrict__ Wout, const float* __restrict__ bout,
                      float* __restrict__ out) {
    int g = blockIdx.x * blockDim.x + threadIdx.x;
    if (g >= NG) return;
    float inv = 1.0f / fmaxf(gcnt[g], 1.0f);
    float s = 0.0f;
    for (int f = 0; f < DIM; ++f) s += gsum[g * DIM + f] * Wout[f];
    out[g] = s * inv + bout[0];
}

extern "C" void kernel_launch(void* const* d_in, const int* in_sizes, int n_in,
                              void* d_out, int out_size, void* d_ws, size_t ws_size,
                              hipStream_t stream) {
    const float* x     = (const float*)d_in[0];
    const int*   ei    = (const int*)d_in[1];
    const int*   batch = (const int*)d_in[2];
    const float* Wl[3] = {(const float*)d_in[3], (const float*)d_in[6], (const float*)d_in[9]};
    const float* bl[3] = {(const float*)d_in[4], (const float*)d_in[7], (const float*)d_in[10]};
    const float* Wr[3] = {(const float*)d_in[5], (const float*)d_in[8], (const float*)d_in[11]};
    const float* Wout  = (const float*)d_in[12];
    const float* bout  = (const float*)d_in[13];
    const int* src = ei;
    const int* dst = ei + NE;

    // workspace carve-up (all chunks 32B-aligned)
    char* ws = (char*)d_ws;
    float*  agg   = (float*)ws;                ws += (size_t)NN * DIM * 4;   // 25.6 MB
    float*  bufA  = (float*)ws;                ws += (size_t)NN * DIM * 4;   // 25.6 MB
    float*  bufB  = (float*)ws;                ws += (size_t)NN * DIM * 4;   // 25.6 MB
    float*  cnt   = (float*)ws;                ws += (size_t)NN * 4;         // 200 KB
    __bf16* wpack = (__bf16*)ws;               ws += (size_t)3 * 32768 * 2;  // 192 KB
    float*  gsum  = (float*)ws;                ws += (size_t)NG * DIM * 4;   // 256 KB
    float*  gcnt  = (float*)ws;

    const int B = 256;

    // degree once (dst is layer-invariant)
    k_zero<<<(NN + B - 1) / B, B, 0, stream>>>(cnt, NN);
    k_degree<<<(NE + B - 1) / B, B, 0, stream>>>(dst, cnt);

    // pre-pack weights of all 3 layers into WMMA fragment order
    for (int l = 0; l < 3; ++l)
        k_pack<<<(32768 + B - 1) / B, B, 0, stream>>>(Wl[l], Wr[l], wpack + l * 32768);

    const float* xin = x;
    float* xout = bufA;
    for (int l = 0; l < 3; ++l) {
        k_zero<<<(NN * DIM + B - 1) / B, B, 0, stream>>>(agg, NN * DIM);
        long long st = (long long)NE * 32;
        k_scatter<<<(unsigned)((st + B - 1) / B), B, 0, stream>>>(xin, src, dst, agg);
        k_sage_gemm<<<NN / 16, 128, 0, stream>>>(xin, agg, cnt, wpack + l * 32768, bl[l], xout);
        xin = xout;
        xout = (l == 0) ? bufB : bufA;
    }
    // xin now holds final node features (bufA)

    k_zero<<<(NG * DIM + B - 1) / B, B, 0, stream>>>(gsum, NG * DIM);
    k_zero<<<(NG + B - 1) / B, B, 0, stream>>>(gcnt, NG);
    long long pt = (long long)NN * 32;
    k_pool<<<(unsigned)((pt + B - 1) / B), B, 0, stream>>>(xin, batch, gsum, gcnt);
    k_out<<<(NG + B - 1) / B, B, 0, stream>>>(gsum, gcnt, Wout, bout, (float*)d_out);
}
